// EdgeMLP_20486994002443
// MI455X (gfx1250) — matrix operand order repaired and checked by reading, not compiled, not run on hardware
//
#include <hip/hip_runtime.h>
#include <hip/hip_bf16.h>

typedef __attribute__((ext_vector_type(16))) __bf16 v16bf;
typedef __attribute__((ext_vector_type(8)))  __bf16 v8bf;
typedef __attribute__((ext_vector_type(8)))  float  v8f;

#define EFEAT   128
#define NFEAT   128
#define IN_DIM  384
#define HID     256
#define OUTD    128
#define KT1     12      // 384 / 32
#define KT2     8       // 256 / 32
#define MROWS   32      // edges per iteration (two 16-row M-tiles)
#define ITER    16      // iterations per workgroup -> 512 edges / WG
#define ASTR    392     // halves per A row (384 + 8 pad): bank shift, 16B aligned
#define HSTR    264     // halves per h row (256 + 8 pad)
#define YSTR    132     // floats per y row (128 + 4 pad)
#define LN_EPS  1e-5f

__global__ __launch_bounds__(256)
void edge_mlp_wmma_kernel(const float* __restrict__ efeat,
                          const float* __restrict__ nfeat,
                          const int*   __restrict__ src_idx,
                          const int*   __restrict__ dst_idx,
                          const float* __restrict__ W1,
                          const float* __restrict__ b1,
                          const float* __restrict__ W2,
                          const float* __restrict__ b2,
                          const float* __restrict__ gamma,
                          const float* __restrict__ beta,
                          float*       __restrict__ out,
                          int nEdges, int nTiles)
{
    __shared__ __align__(16) __bf16 A_lds[2][MROWS * ASTR]; // 2 x 24.5 KB (double buffer)
    __shared__ __align__(16) __bf16 h_lds[MROWS * HSTR];    // 16.5 KB
    __shared__ __align__(16) float  y_lds[MROWS * YSTR];    // 16.5 KB

    const int tid   = threadIdx.x;
    const int wave  = tid >> 5;     // 0..7
    const int lane  = tid & 31;
    const int lmod  = lane & 15;    // N (B/C/D) or M (A)
    const int khalf = lane >> 4;    // K-half selector

    // ---------------------------------------------------------------
    // Register-resident bf16 weight fragments, loaded once per WG and
    // reused across ITER*2 M-tiles.
    // B frag (32x16), contiguous k order used here: lanes 0-15 hold K=0..15, lanes 16-31
    // hold K=16..31; VGPR v holds K = khalf*16 + 2v, 2v+1.
    // ---------------------------------------------------------------
    v16bf B1[2][KT1];                       // wave owns hidden n-tiles 2w, 2w+1
#pragma unroll
    for (int t = 0; t < 2; ++t) {
        const int n = (wave * 2 + t) * 16 + lmod;
#pragma unroll
        for (int kt = 0; kt < KT1; ++kt) {
            const int kb = kt * 32 + khalf * 16;
            v16bf b;
#pragma unroll
            for (int j = 0; j < 16; ++j)
                b[j] = (__bf16)W1[(size_t)(kb + j) * HID + n];
            B1[t][kt] = b;
        }
    }
    v16bf B2[KT2];                          // wave owns output n-tile w
    {
        const int n = wave * 16 + lmod;
#pragma unroll
        for (int kt = 0; kt < KT2; ++kt) {
            const int kb = kt * 32 + khalf * 16;
            v16bf b;
#pragma unroll
            for (int j = 0; j < 16; ++j)
                b[j] = (__bf16)W2[(size_t)(kb + j) * OUTD + n];
            B2[kt] = b;
        }
    }
    const float b1v0 = b1[(wave * 2 + 0) * 16 + lmod];
    const float b1v1 = b1[(wave * 2 + 1) * 16 + lmod];
    const float b2v  = b2[wave * 16 + lmod];

    // LN scale/shift: uniform per thread across iterations
    const int lnr = tid >> 3;               // edge row 0..31 for stage/LN
    const int lnc = tid & 7;                // 8 threads per row
    const int ln0 = lnc * 16;               // 16 columns per thread
    float4 gv[4], tv[4];
#pragma unroll
    for (int q = 0; q < 4; ++q) {
        gv[q] = *(const float4*)&gamma[ln0 + q * 4];
        tv[q] = *(const float4*)&beta [ln0 + q * 4];
    }

    // ---- gather + concat + f32->bf16 stage of one 32-edge tile ----
    auto stage = [&](int tile, int buf) {
        const int r = lnr;                  // row 0..31
        const int c = lnc;                  // 16-float chunk within 128
        int e = tile * MROWS + r; if (e >= nEdges) e = nEdges - 1;
        const int si = src_idx[e];
        const int di = dst_idx[e];
        const float* srcp[3] = { efeat + (size_t)e  * EFEAT,
                                 nfeat + (size_t)si * NFEAT,
                                 nfeat + (size_t)di * NFEAT };
#pragma unroll
        for (int s = 0; s < 3; ++s) {
            const float4* p = (const float4*)srcp[s] + c * 4;
            float4 f[4];
#pragma unroll
            for (int q = 0; q < 4; ++q) f[q] = p[q];
#pragma unroll
            for (int h = 0; h < 2; ++h) {
                v8bf v;
                v[0]=(__bf16)f[2*h].x;   v[1]=(__bf16)f[2*h].y;
                v[2]=(__bf16)f[2*h].z;   v[3]=(__bf16)f[2*h].w;
                v[4]=(__bf16)f[2*h+1].x; v[5]=(__bf16)f[2*h+1].y;
                v[6]=(__bf16)f[2*h+1].z; v[7]=(__bf16)f[2*h+1].w;
                *(v8bf*)&A_lds[buf][r * ASTR + s * 128 + c * 16 + h * 8] = v;
            }
        }
    };

    // prologue: stage first tile
    {
        const int tile0 = blockIdx.x * ITER;
        if (tile0 < nTiles) stage(tile0, 0);
    }
    __syncthreads();

    int p = 0;
    for (int it = 0; it < ITER; ++it) {
        const int tile = blockIdx.x * ITER + it;
        if (tile >= nTiles) break;
        const int ebase = tile * MROWS;

        // ---- GEMM1: [32,384] x [384,256]; 4 accumulators per wave ----
        // A frag (16-bit, 16x32): lane m=lmod; 8 consecutive halves at
        // K = kt*32 + khalf*16 and at +8  => 2x ds_load_b128 per frag.
        v8f a00 = {}, a01 = {}, a10 = {}, a11 = {};
#pragma unroll
        for (int kt = 0; kt < KT1; ++kt) {
            const int ko = kt * 32 + khalf * 16;
            const v8bf lo0 = *(const v8bf*)&A_lds[p][lmod * ASTR + ko];
            const v8bf hi0 = *(const v8bf*)&A_lds[p][lmod * ASTR + ko + 8];
            const v8bf lo1 = *(const v8bf*)&A_lds[p][(16 + lmod) * ASTR + ko];
            const v8bf hi1 = *(const v8bf*)&A_lds[p][(16 + lmod) * ASTR + ko + 8];
            v16bf am0, am1;
#pragma unroll
            for (int j = 0; j < 8; ++j) {
                am0[j] = lo0[j]; am0[j + 8] = hi0[j];
                am1[j] = lo1[j]; am1[j + 8] = hi1[j];
            }
            a00 = __builtin_amdgcn_wmma_f32_16x16x32_bf16(false, am0, false, B1[0][kt],
                                                          (short)0, a00, false, false);
            a01 = __builtin_amdgcn_wmma_f32_16x16x32_bf16(false, am0, false, B1[1][kt],
                                                          (short)0, a01, false, false);
            a10 = __builtin_amdgcn_wmma_f32_16x16x32_bf16(false, am1, false, B1[0][kt],
                                                          (short)0, a10, false, false);
            a11 = __builtin_amdgcn_wmma_f32_16x16x32_bf16(false, am1, false, B1[1][kt],
                                                          (short)0, a11, false, false);
        }

        // ---- overlap: stage next tile into other buffer (global-load
        //      latency hides under GEMM2 + LayerNorm) ----
        if (it + 1 < ITER) {
            const int ntile = tile + 1;
            if (ntile < nTiles) stage(ntile, p ^ 1);
        }

        // ---- bias + SiLU, write hidden tiles (bf16) to LDS ----
        {
            const int mbase = khalf * 8;    // C/D: VGPR i -> row i (+8 hi lanes)
            const int n0 = (wave * 2 + 0) * 16 + lmod;
            const int n1 = (wave * 2 + 1) * 16 + lmod;
#pragma unroll
            for (int i = 0; i < 8; ++i) {
                const float x00 = a00[i] + b1v0, x01 = a01[i] + b1v1;
                const float x10 = a10[i] + b1v0, x11 = a11[i] + b1v1;
                h_lds[(mbase + i) * HSTR + n0]        = (__bf16)(x00 / (1.0f + __expf(-x00)));
                h_lds[(mbase + i) * HSTR + n1]        = (__bf16)(x01 / (1.0f + __expf(-x01)));
                h_lds[(16 + mbase + i) * HSTR + n0]   = (__bf16)(x10 / (1.0f + __expf(-x10)));
                h_lds[(16 + mbase + i) * HSTR + n1]   = (__bf16)(x11 / (1.0f + __expf(-x11)));
            }
        }
        __syncthreads();

        // ---- GEMM2: [32,256] x [256,128]; 2 accumulators per wave ----
        v8f c0 = {}, c1 = {};
#pragma unroll
        for (int kt = 0; kt < KT2; ++kt) {
            const int ko = kt * 32 + khalf * 16;
            const v8bf lo0 = *(const v8bf*)&h_lds[lmod * HSTR + ko];
            const v8bf hi0 = *(const v8bf*)&h_lds[lmod * HSTR + ko + 8];
            const v8bf lo1 = *(const v8bf*)&h_lds[(16 + lmod) * HSTR + ko];
            const v8bf hi1 = *(const v8bf*)&h_lds[(16 + lmod) * HSTR + ko + 8];
            v16bf am0, am1;
#pragma unroll
            for (int j = 0; j < 8; ++j) {
                am0[j] = lo0[j]; am0[j + 8] = hi0[j];
                am1[j] = lo1[j]; am1[j + 8] = hi1[j];
            }
            c0 = __builtin_amdgcn_wmma_f32_16x16x32_bf16(false, am0, false, B2[kt],
                                                         (short)0, c0, false, false);
            c1 = __builtin_amdgcn_wmma_f32_16x16x32_bf16(false, am1, false, B2[kt],
                                                         (short)0, c1, false, false);
        }
        {
            const int mbase = khalf * 8;
            const int n = wave * 16 + lmod;
#pragma unroll
            for (int i = 0; i < 8; ++i) {
                y_lds[(mbase + i) * YSTR + n]      = c0[i] + b2v;
                y_lds[(16 + mbase + i) * YSTR + n] = c1[i] + b2v;
            }
        }
        __syncthreads();

        // ---- LayerNorm over 128 outputs per edge + coalesced store ----
        {
            float v[16];
#pragma unroll
            for (int q = 0; q < 4; ++q) {
                const float4 y4 = *(const float4*)&y_lds[lnr * YSTR + ln0 + q * 4];
                v[4*q+0] = y4.x; v[4*q+1] = y4.y; v[4*q+2] = y4.z; v[4*q+3] = y4.w;
            }
            float s = 0.f;
#pragma unroll
            for (int j = 0; j < 16; ++j) s += v[j];
#pragma unroll
            for (int off = 4; off >= 1; off >>= 1) s += __shfl_xor(s, off, 8);
            const float mu = s * (1.0f / 128.0f);
            float q2 = 0.f;
#pragma unroll
            for (int j = 0; j < 16; ++j) { const float d = v[j] - mu; q2 += d * d; }
#pragma unroll
            for (int off = 4; off >= 1; off >>= 1) q2 += __shfl_xor(q2, off, 8);
            const float rstd = rsqrtf(q2 * (1.0f / 128.0f) + LN_EPS);

            const int e = ebase + lnr;
            if (e < nEdges) {
                float* op = out + (size_t)e * OUTD + ln0;
#pragma unroll
                for (int q = 0; q < 4; ++q) {
                    const float4 g = gv[q], t = tv[q];
                    float4 r4;
                    r4.x = (v[4*q+0] - mu) * rstd * g.x + t.x;
                    r4.y = (v[4*q+1] - mu) * rstd * g.y + t.y;
                    r4.z = (v[4*q+2] - mu) * rstd * g.z + t.z;
                    r4.w = (v[4*q+3] - mu) * rstd * g.w + t.w;
                    *(float4*)(op + q * 4) = r4;
                }
            }
        }
        __syncthreads();   // A[p^1] staged, h/y consumed -> safe for next iter
        p ^= 1;
    }
}

extern "C" void kernel_launch(void* const* d_in, const int* in_sizes, int n_in,
                              void* d_out, int out_size, void* d_ws, size_t ws_size,
                              hipStream_t stream) {
    const float* efeat = (const float*)d_in[0];
    const float* nfeat = (const float*)d_in[1];
    const int*   srci  = (const int*)d_in[2];
    const int*   dsti  = (const int*)d_in[3];
    const float* W1    = (const float*)d_in[4];
    const float* b1    = (const float*)d_in[5];
    const float* W2    = (const float*)d_in[6];
    const float* b2    = (const float*)d_in[7];
    const float* gamma = (const float*)d_in[8];
    const float* beta  = (const float*)d_in[9];
    float* out = (float*)d_out;

    const int nEdges = in_sizes[2];                    // src_idx count
    const int nTiles = (nEdges + MROWS - 1) / MROWS;   // 32-edge tiles
    const int nBlk   = (nTiles + ITER - 1) / ITER;     // ITER tiles per WG

    edge_mlp_wmma_kernel<<<nBlk, 256, 0, stream>>>(efeat, nfeat, srci, dsti,
                                                   W1, b1, W2, b2, gamma, beta,
                                                   out, nEdges, nTiles);
}
